// AttentionBlock_45792941310373
// MI455X (gfx1250) — compile-verified, matrix operand order checked by reading, not run on hardware
//
#include <hip/hip_runtime.h>
#include <hip/hip_bf16.h>

typedef _Float16 half_t;
typedef __attribute__((ext_vector_type(16))) _Float16 v16h;
typedef __attribute__((ext_vector_type(8)))  _Float16 v8h;
typedef __attribute__((ext_vector_type(8)))  float    v8f;

union V16 { v16h v; v8h h[2]; };

// Problem constants (match reference)
constexpr int BB   = 2;
constexpr int TT   = 2048;
constexpr int DM   = 1024;
constexpr int HN   = 16;
constexpr int DH   = 64;
constexpr int QKVD = 3072;
constexpr int MROW = BB * TT;          // 4096
constexpr int WIN  = 128;
constexpr float SM_SCALE = 0.125f;     // 1/sqrt(64)
constexpr float MASKED = -3.0e38f;
constexpr float MINIT  = -1.0e30f;

// Workspace layout (bytes)
constexpr size_t OFF_QKVW = 0;
constexpr size_t SZ_QKVW  = (size_t)QKVD * DM * 2;
constexpr size_t OFF_OUTW = OFF_QKVW + SZ_QKVW;
constexpr size_t SZ_OUTW  = (size_t)DM * DM * 2;
constexpr size_t OFF_T    = OFF_OUTW + SZ_OUTW;
constexpr size_t SZ_T     = (size_t)MROW * DM * 2;
constexpr size_t OFF_QKVF = OFF_T + SZ_T;
constexpr size_t SZ_QKVF  = (size_t)MROW * QKVD * 4;
constexpr size_t OFF_ATTN = OFF_QKVF;                 // alias: qkv f32 dead after RoPE
constexpr size_t OFF_Q    = OFF_QKVF + SZ_QKVF;
constexpr size_t SZ_QKH   = (size_t)BB * HN * TT * DH * 2;
constexpr size_t OFF_K    = OFF_Q + SZ_QKH;
constexpr size_t OFF_V    = OFF_K + SZ_QKH;

// ---------------------------------------------------------------------------
// fragment helpers
// ---------------------------------------------------------------------------
__device__ __forceinline__ void ldf(V16& f, const half_t* p) {
  f.h[0] = *(const v8h*)(p);
  f.h[1] = *(const v8h*)(p + 16);
}

// ---------------------------------------------------------------------------
// 1) RMSNorm: x (f32, row=1024) -> t (f16)
// ---------------------------------------------------------------------------
__global__ __launch_bounds__(256) void rmsnorm_kernel(
    const float* __restrict__ x, const float* __restrict__ scale,
    half_t* __restrict__ t) {
  const int row = blockIdx.x;
  const float* xr = x + (size_t)row * DM;
  float ss = 0.f;
  for (int i = threadIdx.x; i < DM; i += 256) { float v = xr[i]; ss += v * v; }
  #pragma unroll
  for (int o = 16; o > 0; o >>= 1) ss += __shfl_xor(ss, o, 32);
  __shared__ float red[8];
  if ((threadIdx.x & 31) == 0) red[threadIdx.x >> 5] = ss;
  __syncthreads();
  float tot = 0.f;
  #pragma unroll
  for (int i = 0; i < 8; ++i) tot += red[i];
  const float inv = rsqrtf(tot * (1.0f / DM) + 1e-5f);
  for (int i = threadIdx.x; i < DM; i += 256)
    t[(size_t)row * DM + i] = (half_t)(xr[i] * inv * scale[i]);
}

// ---------------------------------------------------------------------------
// 2) f32 -> f16 weight conversion
// ---------------------------------------------------------------------------
__global__ __launch_bounds__(256) void f32_to_f16_kernel(
    const float* __restrict__ src, half_t* __restrict__ dst, int n) {
  int i = blockIdx.x * blockDim.x + threadIdx.x;
  const int stride = gridDim.x * blockDim.x;
  for (; i < n; i += stride) dst[i] = (half_t)src[i];
}

// ---------------------------------------------------------------------------
// 3/6) WMMA GEMM: C[M,N] = A[M,K](f16) * W[N,K](f16)^T + bias (+ residual)
//      One wave computes a 32x64 slab (2 A tiles x 4 W tiles = 8 WMMA/step),
//      with explicit double buffering so loads overlap WMMA issue.
//      Requires K % 64 == 0.
// ---------------------------------------------------------------------------
__device__ __forceinline__ void mma8(v8f acc[2][4], const V16 a[2], const V16 b[4]) {
  #pragma unroll
  for (int i = 0; i < 2; ++i)
    #pragma unroll
    for (int j = 0; j < 4; ++j)
      acc[i][j] = __builtin_amdgcn_wmma_f32_16x16x32_f16(
          false, a[i].v, false, b[j].v, (short)0, acc[i][j], false, false);
}

__device__ __forceinline__ void load_step(
    V16 a[2], V16 b[4],
    const half_t* A0, const half_t* A1,
    const half_t* W0, const half_t* W1,
    const half_t* W2, const half_t* W3, int k) {
  ldf(a[0], A0 + k);
  ldf(a[1], A1 + k);
  ldf(b[0], W0 + k);
  ldf(b[1], W1 + k);
  ldf(b[2], W2 + k);
  ldf(b[3], W3 + k);
}

__global__ __launch_bounds__(256) void gemm_wmma_kernel(
    const half_t* __restrict__ A, const half_t* __restrict__ W,
    const float* __restrict__ bias, const float* __restrict__ residual,
    float* __restrict__ C, int M, int N, int K) {
  const int lane = threadIdx.x & 31;
  const int wid  = (blockIdx.x * blockDim.x + threadIdx.x) >> 5;
  const int ntn  = N >> 6;             // 64-wide col tiles
  const int ntm  = M >> 5;             // 32-high row tiles
  if (wid >= ntm * ntn) return;
  const int tm = wid / ntn, tn = wid % ntn;  // consecutive waves share A rows
  const int ko = (lane >= 16) ? 8 : 0;
  const int ln = lane & 15;

  const half_t* A0 = A + (size_t)(tm * 32 + ln) * K + ko;
  const half_t* A1 = A0 + (size_t)16 * K;
  const half_t* W0 = W + (size_t)(tn * 64 + ln) * K + ko;
  const half_t* W1 = W0 + (size_t)16 * K;
  const half_t* W2 = W0 + (size_t)32 * K;
  const half_t* W3 = W0 + (size_t)48 * K;

  v8f acc[2][4] = {};
  V16 pa[2], pb[4], qa[2], qb[4];

  // software pipeline: steps of 32 in K, ping-pong buffers
  load_step(pa, pb, A0, A1, W0, W1, W2, W3, 0);
  int k = 32;
  for (; k < K - 32; k += 64) {
    load_step(qa, qb, A0, A1, W0, W1, W2, W3, k);
    mma8(acc, pa, pb);
    load_step(pa, pb, A0, A1, W0, W1, W2, W3, k + 32);
    mma8(acc, qa, qb);
  }
  load_step(qa, qb, A0, A1, W0, W1, W2, W3, k);  // k == K-32
  mma8(acc, pa, pb);
  mma8(acc, qa, qb);

  #pragma unroll
  for (int i = 0; i < 2; ++i) {
    #pragma unroll
    for (int j = 0; j < 4; ++j) {
      const int col = tn * 64 + j * 16 + ln;
      const float bv = bias[col];
      #pragma unroll
      for (int r = 0; r < 8; ++r) {
        const int row = tm * 32 + i * 16 + r + ko;  // lanes>=16 hold M=8+r
        float v = acc[i][j][r] + bv;
        if (residual) v += residual[(size_t)row * N + col];
        C[(size_t)row * N + col] = v;
      }
    }
  }
}

// ---------------------------------------------------------------------------
// 4) RoPE + layout: qkv f32 -> Q,K (b,h,t,d) f16 and V^T (b,h,d,t) f16
// ---------------------------------------------------------------------------
__global__ __launch_bounds__(256) void rope_kernel(
    const float* __restrict__ qkv, half_t* __restrict__ Q,
    half_t* __restrict__ Kr, half_t* __restrict__ Vt) {
  const int idx = blockIdx.x * blockDim.x + threadIdx.x;  // exact grid
  const int d = idx & 63;
  const int h = (idx >> 6) & (HN - 1);
  const int t = (idx >> 10) & (TT - 1);
  const int b = idx >> 21;
  const int p = d & 31;
  // inv_freq = 10000^(-p/32); ln(10000) = 9.210340371976184
  const float ang = (float)t * __expf(-(float)p * (9.210340372f / 32.0f));
  const float c = cosf(ang), sn = sinf(ang);
  const size_t base = (size_t)(b * TT + t) * QKVD;
  const float q1 = qkv[base + h * 64 + p];
  const float q2 = qkv[base + h * 64 + p + 32];
  const float k1 = qkv[base + DM + h * 64 + p];
  const float k2 = qkv[base + DM + h * 64 + p + 32];
  const float qv = (d < 32) ? (q1 * c - q2 * sn) : (q2 * c + q1 * sn);
  const float kv = (d < 32) ? (k1 * c - k2 * sn) : (k2 * c + k1 * sn);
  const size_t bh = (size_t)(b * HN + h);
  Q [(bh * TT + t) * DH + d] = (half_t)qv;
  Kr[(bh * TT + t) * DH + d] = (half_t)kv;
  Vt[(bh * DH + d) * TT + t] = (half_t)qkv[base + 2 * DM + h * 64 + d];
}

// ---------------------------------------------------------------------------
// 5) Sliding-window attention with sink. One wave per (b,h, 16-query block).
//    Scores computed transposed (S^T = K*Q^T) so softmax is lane-local and
//    P^T registers are directly the B-operand of the O^T = V^T * P^T WMMA.
// ---------------------------------------------------------------------------
__global__ __launch_bounds__(256) void attn_kernel(
    const half_t* __restrict__ Q, const half_t* __restrict__ Kr,
    const half_t* __restrict__ Vt, const int* __restrict__ amask,
    const float* __restrict__ sinks, half_t* __restrict__ attn) {
  const int lane = threadIdx.x & 31;
  const int wid  = (blockIdx.x * blockDim.x + threadIdx.x) >> 5;
  const int QT   = TT / 16;
  const int qt   = wid % QT;
  const int bh   = wid / QT;
  const int h    = bh % HN, b = bh / HN;
  const int qbase = qt * 16;
  const int ln = lane & 15;
  const int ko = (lane >= 16) ? 8 : 0;
  const int iq = qbase + ln;                       // this lane's query index
  const half_t* Qh = Q  + (size_t)bh * TT * DH;
  const half_t* Kh = Kr + (size_t)bh * TT * DH;
  const half_t* Vh = Vt + (size_t)bh * DH * TT;
  const int* am = amask + b * TT;

  // Q as B-fragments for the two K=32 chunks of head_dim
  V16 qb0, qb1;
  {
    const half_t* qp = Qh + (size_t)(qbase + ln) * DH;
    ldf(qb0, qp + ko);
    ldf(qb1, qp + 32 + ko);
  }

  v8f acc[4] = {};
  float m = MINIT, l = 0.f;
  int kstart = qbase - (WIN - 1); if (kstart < 0) kstart = 0;
  kstart &= ~31;
  const int kend = qbase + 16;

  for (int jc = kstart; jc < kend; jc += 32) {
    v8f s[2];
    #pragma unroll
    for (int t2 = 0; t2 < 2; ++t2) {
      int jrow = jc + t2 * 16 + ln;
      if (jrow > TT - 1) jrow = TT - 1;            // clamp (masked below)
      const half_t* kp = Kh + (size_t)jrow * DH;
      V16 a0, a1;
      ldf(a0, kp + ko);
      ldf(a1, kp + 32 + ko);
      v8f z = {};
      z = __builtin_amdgcn_wmma_f32_16x16x32_f16(false, a0.v, false, qb0.v,
                                                 (short)0, z, false, false);
      z = __builtin_amdgcn_wmma_f32_16x16x32_f16(false, a1.v, false, qb1.v,
                                                 (short)0, z, false, false);
      s[t2] = z;
    }
    // mask + per-query (lane-local) running max
    float tl = MASKED;
    #pragma unroll
    for (int t2 = 0; t2 < 2; ++t2) {
      #pragma unroll
      for (int r = 0; r < 8; ++r) {
        const int j = jc + t2 * 16 + r + ko;       // D layout key index
        const int jcl = (j < TT - 1) ? j : TT - 1;
        const bool ok = (j <= iq) && (j + WIN > iq) && (j < TT) && (am[jcl] != 0);
        const float v = ok ? s[t2][r] * SM_SCALE : MASKED;
        s[t2][r] = v;
        tl = fmaxf(tl, v);
      }
    }
    tl = fmaxf(tl, __shfl_xor(tl, 16, 32));        // partner half-lane has other 8 keys
    const float mnew = fmaxf(m, tl);
    const float co = __expf(m - mnew);
    l *= co;
    #pragma unroll
    for (int dt = 0; dt < 4; ++dt) acc[dt] = acc[dt] * co;

    // P^T -> exactly the B operand layout of the PV WMMA
    V16 pbf;
    float psum = 0.f;
    #pragma unroll
    for (int r = 0; r < 8; ++r) {
      const float p0 = __expf(s[0][r] - mnew);
      const float p1 = __expf(s[1][r] - mnew);
      psum += p0 + p1;
      pbf.h[0][r] = (half_t)p0;
      pbf.h[1][r] = (half_t)p1;
    }
    l += psum + __shfl_xor(psum, 16, 32);
    m = mnew;

    // O^T += V^T * P^T  (4 d-tiles of 16)
    #pragma unroll
    for (int dt = 0; dt < 4; ++dt) {
      const half_t* vp = Vh + (size_t)(dt * 16 + ln) * TT + jc + ko;
      V16 va;
      ldf(va, vp);
      acc[dt] = __builtin_amdgcn_wmma_f32_16x16x32_f16(
          false, va.v, false, pbf.v, (short)0, acc[dt], false, false);
    }
  }

  // fold in the sink column, normalize, store f16 (contiguous v8h per tile)
  const float snk = sinks[h];
  const float m2 = fmaxf(m, snk);
  const float co = __expf(m - m2);
  const float ltot = l * co + __expf(snk - m2);
  const float factor = co / ltot;
  half_t* op = attn + (size_t)(b * TT + qbase + ln) * DM + h * 64;
  #pragma unroll
  for (int dt = 0; dt < 4; ++dt) {
    v8h o;
    #pragma unroll
    for (int r = 0; r < 8; ++r) o[r] = (half_t)(acc[dt][r] * factor);
    *(v8h*)(op + dt * 16 + ko) = o;
  }
}

// ---------------------------------------------------------------------------
extern "C" void kernel_launch(void* const* d_in, const int* in_sizes, int n_in,
                              void* d_out, int out_size, void* d_ws, size_t ws_size,
                              hipStream_t stream) {
  const float* x      = (const float*)d_in[0];
  const int*   amask  = (const int*)d_in[1];
  const float* nscale = (const float*)d_in[2];
  const float* qkv_w  = (const float*)d_in[3];
  const float* qkv_b  = (const float*)d_in[4];
  const float* out_w  = (const float*)d_in[5];
  const float* out_b  = (const float*)d_in[6];
  const float* sinks  = (const float*)d_in[7];
  float* out = (float*)d_out;
  char*  ws  = (char*)d_ws;

  half_t* qkvw_h = (half_t*)(ws + OFF_QKVW);
  half_t* outw_h = (half_t*)(ws + OFF_OUTW);
  half_t* t_h    = (half_t*)(ws + OFF_T);
  float*  qkvf   = (float*)(ws + OFF_QKVF);
  half_t* attn_h = (half_t*)(ws + OFF_ATTN);
  half_t* Qh     = (half_t*)(ws + OFF_Q);
  half_t* Kh     = (half_t*)(ws + OFF_K);
  half_t* Vth    = (half_t*)(ws + OFF_V);

  rmsnorm_kernel<<<MROW, 256, 0, stream>>>(x, nscale, t_h);
  f32_to_f16_kernel<<<2048, 256, 0, stream>>>(qkv_w, qkvw_h, QKVD * DM);
  f32_to_f16_kernel<<<1024, 256, 0, stream>>>(out_w, outw_h, DM * DM);

  // QKV GEMM: 4096x3072x1024, 32x64 per wave -> 6144 waves -> 768 blocks
  gemm_wmma_kernel<<<768, 256, 0, stream>>>(t_h, qkvw_h, qkv_b, nullptr,
                                            qkvf, MROW, QKVD, DM);

  rope_kernel<<<(BB * TT * HN * DH) / 256, 256, 0, stream>>>(qkvf, Qh, Kh, Vth);

  // attention: B*H*(T/16) = 4096 waves -> 512 blocks
  attn_kernel<<<512, 256, 0, stream>>>(Qh, Kh, Vth, amask, sinks, attn_h);

  // out proj + bias + residual: 4096x1024x1024, 32x64/wave -> 2048 waves -> 256 blocks
  gemm_wmma_kernel<<<256, 256, 0, stream>>>(attn_h, outw_h, out_b, x,
                                            out, MROW, DM, DM);
}
